// LSTM2MLP_67087389164061
// MI455X (gfx1250) — compile-verified, matrix operand order checked by reading, not run on hardware
//
#include <hip/hip_runtime.h>
#include <hip/hip_bf16.h>
#include <math.h>

// ---------------------------------------------------------------------------
// 2-layer biLSTM + linear head for MI455X (gfx1250, wave32, WMMA).
// bf16 storage / f32 WMMA accumulation; async global->LDS double-buffering.
// ---------------------------------------------------------------------------

typedef __bf16 bf16;
typedef __attribute__((ext_vector_type(16))) __bf16 v16bf;
typedef __attribute__((ext_vector_type(8)))  __bf16 v8bf;
typedef __attribute__((ext_vector_type(8)))  float  v8f;

#define DEVINL __device__ __forceinline__

constexpr int Bm   = 64;
constexpr int Sm   = 512;
constexpr int Em   = 300;
constexpr int Hm   = 256;
constexpr int G4H  = 4 * Hm;   // 1024 gate columns
constexpr int E_PAD = 320;     // 300 padded to multiple of 32
constexpr int D1   = 2 * Hm;   // 512 (bi-directional concat)

DEVINL bf16 f2bf(float f) {
  unsigned u = __builtin_bit_cast(unsigned, f);
  u += 0x7FFFu + ((u >> 16) & 1u);               // round-to-nearest-even
  unsigned short s = (unsigned short)(u >> 16);
  return __builtin_bit_cast(bf16, s);
}
DEVINL float bf2f(bf16 b) {
  unsigned short s = __builtin_bit_cast(unsigned short, b);
  unsigned u = ((unsigned)s) << 16;
  return __builtin_bit_cast(float, u);
}

// CDNA5 16-bit A/B fragment layout: lanes 0-15 hold K=[0..7],[16..23];
// lanes 16-31 hold K=[8..15],[24..31]. Two 16B loads per lane.
DEVINL v16bf load_frag(const bf16* rowptr, int lane) {
  const int h = (lane >> 4) << 3;  // 0 or 8
  union { v16bf v; v8bf p[2]; } u;
  u.p[0] = *(const v8bf*)(rowptr + h);
  u.p[1] = *(const v8bf*)(rowptr + 16 + h);
  return u.v;
}

// Async global -> LDS copy of one 16B chunk (per-lane addresses).
// VDST = 32-bit LDS byte address (low 32 bits of the generic pointer),
// VADDR = 64-bit global address. Tracked by ASYNCcnt.
DEVINL void async_copy_b128(const void* gptr, const void* lptr) {
  unsigned lds = (unsigned)(unsigned long long)lptr;
  unsigned long long g = (unsigned long long)gptr;
  asm volatile("global_load_async_to_lds_b128 %0, %1, off"
               :: "v"(lds), "v"(g) : "memory");
}

// ---------------------------------------------------------------------------
// K1: embedding gather + cast to bf16, K padded 300 -> 320 with zeros.
// ---------------------------------------------------------------------------
__global__ void __launch_bounds__(256)
embed_gather_kernel(const int* __restrict__ ids, const float* __restrict__ emb,
                    bf16* __restrict__ X0) {
  long long idx = (long long)blockIdx.x * 256 + threadIdx.x;
  int col = (int)(idx % E_PAD);
  int bs  = (int)(idx / E_PAD);
  float v = 0.f;
  if (col < Em) v = emb[(long long)ids[bs] * Em + col];
  X0[idx] = f2bf(v);
}

// ---------------------------------------------------------------------------
// K2: weight cast to bf16 [N,Kp] (zero-padded K) + bias sums.
// ---------------------------------------------------------------------------
__global__ void __launch_bounds__(256)
cast_weight_kernel(const float* __restrict__ W, bf16* __restrict__ Wb,
                   int K, int Kp, long long total) {
  long long idx = (long long)blockIdx.x * 256 + threadIdx.x;
  if (idx >= total) return;
  int col = (int)(idx % Kp);
  long long row = idx / Kp;
  Wb[idx] = f2bf(col < K ? W[row * K + col] : 0.f);
}

__global__ void __launch_bounds__(256)
bias_sum_kernel(const float* __restrict__ a, const float* __restrict__ b,
                float* __restrict__ dst) {
  int i = blockIdx.x * 256 + threadIdx.x;
  if (i < G4H) dst[i] = a[i] + b[i];
}

// ---------------------------------------------------------------------------
// K3: gate projection GEMM with async double-buffered LDS staging.
// GX[s*B + b][n] = sum_k A[m][k] * W[n][k] + bias[n],   m = b*S + s
// Block: 256 threads (8 waves); tile 128M x 64N, K-step 32.
// LDS: A-tile 2x8KB, B-tile 2x4KB. Each thread issues 3 async b128/stage
// (all unconditional -- EXEC stays all-ones, as WMMA requires).
// ---------------------------------------------------------------------------
__global__ void __launch_bounds__(256)
gemm_gates_kernel(const bf16* __restrict__ A, const bf16* __restrict__ W,
                  const float* __restrict__ bias, bf16* __restrict__ GX, int Kp) {
  const int tid  = threadIdx.x;
  const int lane = tid & 31;
  const int wave = tid >> 5;                         // 0..7
  const int mBase = blockIdx.y * 128;
  const int nBase = blockIdx.x * 64;
  const int mloc  = (wave >> 1) * 32;                // wave-local M offset
  const int nloc  = (wave & 1) * 32;                 // wave-local N offset
  const int lr    = lane & 15;

  __shared__ __attribute__((aligned(16))) bf16 Abuf[2][128 * 32];  // 2 x 8KB
  __shared__ __attribute__((aligned(16))) bf16 Bbuf[2][64 * 32];   // 2 x 4KB

  // stage one K-tile: A = 512 x 16B chunks (2/thread), B = 256 x 16B (1/thread)
  auto stage = [&](int buf, int kk) {
    {                                                // A tile, chunk = tid
      int row = tid >> 2, co = (tid & 3) * 8;
      async_copy_b128(A + (size_t)(mBase + row) * Kp + kk + co,
                      &Abuf[buf][row * 32 + co]);
    }
    {                                                // A tile, chunk = tid+256
      int row = (tid + 256) >> 2, co = (tid & 3) * 8;
      async_copy_b128(A + (size_t)(mBase + row) * Kp + kk + co,
                      &Abuf[buf][row * 32 + co]);
    }
    {                                                // B tile
      int row = tid >> 2, co = (tid & 3) * 8;
      async_copy_b128(W + (size_t)(nBase + row) * Kp + kk + co,
                      &Bbuf[buf][row * 32 + co]);
    }
  };

  v8f acc[2][2];
  v8f zero = {};
#pragma unroll
  for (int i = 0; i < 2; ++i)
#pragma unroll
    for (int j = 0; j < 2; ++j) acc[i][j] = zero;

  stage(0, 0);
  int buf = 0;
  for (int kk = 0; kk < Kp; kk += 32) {
    if (kk + 32 < Kp) {
      stage(buf ^ 1, kk + 32);                       // overlap next copy
      asm volatile("s_wait_asynccnt 0x3" ::: "memory");  // current tile done
    } else {
      asm volatile("s_wait_asynccnt 0x0" ::: "memory");
    }
    __syncthreads();                                 // all waves' copies visible

    v16bf a0 = load_frag(&Abuf[buf][(mloc + lr) * 32], lane);
    v16bf a1 = load_frag(&Abuf[buf][(mloc + 16 + lr) * 32], lane);
    v16bf b0 = load_frag(&Bbuf[buf][(nloc + lr) * 32], lane);
    v16bf b1 = load_frag(&Bbuf[buf][(nloc + 16 + lr) * 32], lane);
    acc[0][0] = __builtin_amdgcn_wmma_f32_16x16x32_bf16(false, a0, false, b0, (short)0, acc[0][0], false, false);
    acc[0][1] = __builtin_amdgcn_wmma_f32_16x16x32_bf16(false, a0, false, b1, (short)0, acc[0][1], false, false);
    acc[1][0] = __builtin_amdgcn_wmma_f32_16x16x32_bf16(false, a1, false, b0, (short)0, acc[1][0], false, false);
    acc[1][1] = __builtin_amdgcn_wmma_f32_16x16x32_bf16(false, a1, false, b1, (short)0, acc[1][1], false, false);

    __syncthreads();                                 // done reading this buffer
    buf ^= 1;
  }

  // Store with [S][B][4H] permutation (scan consumes per-timestep slabs).
#pragma unroll
  for (int ti = 0; ti < 2; ++ti) {
#pragma unroll
    for (int tj = 0; tj < 2; ++tj) {
      const int mt = mBase + mloc + ti * 16 + ((lane >> 4) << 3);
      const int n  = nBase + nloc + tj * 16 + (lane & 15);
      const float bia = bias[n];
#pragma unroll
      for (int v = 0; v < 8; ++v) {
        int m = mt + v;
        int b = m >> 9;       // m / S
        int s = m & 511;      // m % S
        GX[((size_t)s * Bm + b) * G4H + n] = f2bf(acc[ti][tj][v] + bia);
      }
    }
  }
}

// ---------------------------------------------------------------------------
// K4: persistent biLSTM scan. grid=2 (blockIdx.x = direction), 512 threads
// (16 waves). h: bf16 [64,256] in static LDS; c: f32 in registers.
// Per-wave private double-buffered gate stage in 256KB *dynamic* LDS
// (288KB total -- exploits the 320KB CDNA5 WGP LDS), filled by
// global_load_async_to_lds_b128 one timestep ahead (ASYNCcnt pipelined).
// Wave w owns hidden cols [16w,16w+16): 4 M-tiles x 4 gates, K=256
// -> 128 v_wmma per step.
// ---------------------------------------------------------------------------
__global__ void __launch_bounds__(512)
lstm_scan_kernel(const bf16* __restrict__ GXf, const bf16* __restrict__ GXb,
                 const bf16* __restrict__ WHHf, const bf16* __restrict__ WHHb,
                 bf16* __restrict__ OUT /* [B][S][2H] */) {
  const int dir   = blockIdx.x;
  const bf16* GX  = dir ? GXb  : GXf;
  const bf16* WHH = dir ? WHHb : WHHf;
  const int lane  = threadIdx.x & 31;
  const int wave  = threadIdx.x >> 5;      // 0..15
  const int n0    = wave << 4;             // hidden column base
  const int halfM = (lane >> 4) << 3;      // 0 or 8 (C-matrix row half)
  const int lr    = lane & 15;

  __shared__ __attribute__((aligned(32))) bf16 hbuf[Bm * Hm];   // 32 KB
  extern __shared__ bf16 gstage[];   // [2][16 waves][64 rows * 64 cols] bf16

  for (int i = threadIdx.x; i < Bm * Hm; i += 512) hbuf[i] = f2bf(0.f);

  // Stage this wave's 64x64 gate sub-slab (cols {g*256+n0 .. +16} x 4 gates)
  // for timestep s_t: 512 x 16B chunks over 32 lanes = 16 async issues/wave.
  auto stage_gx = [&](int buf, int s_t) {
    const bf16* src = GX + (size_t)s_t * Bm * G4H;
    bf16* dst = gstage + (size_t)(buf * 16 + wave) * (64 * 64);
#pragma unroll
    for (int c = 0; c < 16; ++c) {
      int chunk = lane + c * 32;           // 0..511
      int row   = chunk >> 3;              // batch row 0..63
      int c64   = (chunk & 7) * 8;         // col-in-slab 0,8,..,56
      int gate  = c64 >> 4;
      int col   = gate * Hm + n0 + (c64 & 15);
      async_copy_b128(src + (size_t)row * G4H + col, dst + row * 64 + c64);
    }
  };

  v8f zero = {};
  v8f cst[4];
#pragma unroll
  for (int m = 0; m < 4; ++m) cst[m] = zero;
  __syncthreads();

  stage_gx(0, dir ? (Sm - 1) : 0);
  int buf = 0;
  float hval[4][8];

  for (int t = 0; t < Sm; ++t) {
    const int s = dir ? (Sm - 1 - t) : t;

    if (t + 1 < Sm) {
      stage_gx(buf ^ 1, dir ? (Sm - 2 - t) : (t + 1));   // prefetch next step
      asm volatile("s_wait_asynccnt 0x10" ::: "memory"); // this step's 16 done
    } else {
      asm volatile("s_wait_asynccnt 0x0" ::: "memory");
    }

    // Init gate accumulators from the wave-private LDS stage.
    const bf16* gsl = gstage + (size_t)(buf * 16 + wave) * (64 * 64);
    v8f acc[4][4];
#pragma unroll
    for (int m = 0; m < 4; ++m) {
#pragma unroll
      for (int g = 0; g < 4; ++g) {
        const bf16* p = gsl + (m * 16 + halfM) * 64 + g * 16 + lr;
        v8f a;
#pragma unroll
        for (int v = 0; v < 8; ++v) a[v] = bf2f(p[v * 64]);
        acc[m][g] = a;
      }
    }

    // Recurrent GEMM: gates += h @ w_hh^T  (K = 256, 8 k-tiles).
    for (int kt = 0; kt < Hm; kt += 32) {
      v16bf av[4];
#pragma unroll
      for (int m = 0; m < 4; ++m)
        av[m] = load_frag(hbuf + (size_t)(m * 16 + lr) * Hm + kt, lane);
#pragma unroll
      for (int g = 0; g < 4; ++g) {
        v16bf bv = load_frag(WHH + (size_t)(g * Hm + n0 + lr) * Hm + kt, lane);
#pragma unroll
        for (int m = 0; m < 4; ++m)
          acc[m][g] = __builtin_amdgcn_wmma_f32_16x16x32_bf16(
              false, av[m], false, bv, (short)0, acc[m][g], false, false);
      }
    }

    // LSTM cell update (gate order i,f,g,o).
#pragma unroll
    for (int m = 0; m < 4; ++m) {
#pragma unroll
      for (int v = 0; v < 8; ++v) {
        float ig = 1.f / (1.f + __expf(-acc[m][0][v]));
        float fg = 1.f / (1.f + __expf(-acc[m][1][v]));
        float gg = tanhf(acc[m][2][v]);
        float og = 1.f / (1.f + __expf(-acc[m][3][v]));
        float c  = fg * cst[m][v] + ig * gg;
        cst[m][v] = c;
        hval[m][v] = og * tanhf(c);
      }
    }

    __syncthreads();   // all waves finished reading hbuf for this step
#pragma unroll
    for (int m = 0; m < 4; ++m) {
      const int brow0 = m * 16 + halfM;
#pragma unroll
      for (int v = 0; v < 8; ++v) {
        bf16 hb = f2bf(hval[m][v]);
        int br = brow0 + v;
        hbuf[(size_t)br * Hm + n0 + lr] = hb;
        OUT[((size_t)br * Sm + s) * D1 + dir * Hm + n0 + lr] = hb;
      }
    }
    __syncthreads();   // new h visible before next step's reads
    buf ^= 1;
  }
}

// ---------------------------------------------------------------------------
// K5: head. scores = sigmoid(mean_dir . w_lin + b_lin); masked MSE loss.
// ---------------------------------------------------------------------------
__global__ void zero_loss_kernel(float* out) {
  if (threadIdx.x == 0 && blockIdx.x == 0) out[(size_t)Bm * Sm] = 0.f;
}

__global__ void __launch_bounds__(256)
head_loss_kernel(const bf16* __restrict__ H2, const float* __restrict__ w_lin,
                 const float* __restrict__ b_lin, const float* __restrict__ labels,
                 const long long* __restrict__ masks, float* __restrict__ out) {
  const int b = blockIdx.x;
  __shared__ float serr[256];
  __shared__ float smask[256];
  float errsum = 0.f, masksum = 0.f;
  for (int s = threadIdx.x; s < Sm; s += 256) {
    const bf16* row = H2 + ((size_t)b * Sm + s) * D1;
    float dot = 0.f;
    for (int h = 0; h < Hm; h += 8) {
      v8bf vf = *(const v8bf*)(row + h);
      v8bf vb = *(const v8bf*)(row + Hm + h);
#pragma unroll
      for (int j = 0; j < 8; ++j)
        dot += 0.5f * (bf2f(vf[j]) + bf2f(vb[j])) * w_lin[h + j];
    }
    float score = 1.f / (1.f + __expf(-(dot + b_lin[0])));
    out[(size_t)b * Sm + s] = score;
    float msk = (float)masks[(size_t)b * Sm + s];
    float d   = score - labels[(size_t)b * Sm + s];
    errsum  += d * d * msk;
    masksum += msk;
  }
  serr[threadIdx.x]  = errsum;
  smask[threadIdx.x] = masksum;
  __syncthreads();
  for (int off = 128; off > 0; off >>= 1) {
    if (threadIdx.x < off) {
      serr[threadIdx.x]  += serr[threadIdx.x + off];
      smask[threadIdx.x] += smask[threadIdx.x + off];
    }
    __syncthreads();
  }
  if (threadIdx.x == 0)
    atomicAdd(out + (size_t)Bm * Sm, (serr[0] / smask[0]) * (1.f / Bm));
}

// ---------------------------------------------------------------------------
// Host: deterministic launch sequence, all on `stream`.
// ---------------------------------------------------------------------------
extern "C" void kernel_launch(void* const* d_in, const int* in_sizes, int n_in,
                              void* d_out, int out_size, void* d_ws, size_t ws_size,
                              hipStream_t stream) {
  (void)in_sizes; (void)n_in; (void)out_size; (void)ws_size;

  const int*       ids    = (const int*)d_in[0];
  const float*     labels = (const float*)d_in[1];
  const long long* masks  = (const long long*)d_in[2];
  const float*     emb    = (const float*)d_in[3];
  const float* w_ih[4] = { (const float*)d_in[4],  (const float*)d_in[8],
                           (const float*)d_in[12], (const float*)d_in[16] };
  const float* w_hh[4] = { (const float*)d_in[5],  (const float*)d_in[9],
                           (const float*)d_in[13], (const float*)d_in[17] };
  const float* b_ih[4] = { (const float*)d_in[6],  (const float*)d_in[10],
                           (const float*)d_in[14], (const float*)d_in[18] };
  const float* b_hh[4] = { (const float*)d_in[7],  (const float*)d_in[11],
                           (const float*)d_in[15], (const float*)d_in[19] };
  const float* w_lin = (const float*)d_in[20];
  const float* b_lin = (const float*)d_in[21];
  float* out = (float*)d_out;

  // workspace carve-up (256B aligned slabs)
  char* ws = (char*)d_ws;
  size_t off = 0;
  auto alloc = [&](size_t bytes) -> void* {
    void* p = ws + off;
    off += (bytes + 255) & ~(size_t)255;
    return p;
  };
  bf16* X0    = (bf16*)alloc((size_t)Bm * Sm * E_PAD * 2);   // 21.0 MB
  bf16* Wi0f  = (bf16*)alloc((size_t)G4H * E_PAD * 2);
  bf16* Wi0b  = (bf16*)alloc((size_t)G4H * E_PAD * 2);
  bf16* Wi1f  = (bf16*)alloc((size_t)G4H * D1 * 2);
  bf16* Wi1b  = (bf16*)alloc((size_t)G4H * D1 * 2);
  bf16* Wh0f  = (bf16*)alloc((size_t)G4H * Hm * 2);
  bf16* Wh0b  = (bf16*)alloc((size_t)G4H * Hm * 2);
  bf16* Wh1f  = (bf16*)alloc((size_t)G4H * Hm * 2);
  bf16* Wh1b  = (bf16*)alloc((size_t)G4H * Hm * 2);
  float* Bs0f = (float*)alloc((size_t)G4H * 4);
  float* Bs0b = (float*)alloc((size_t)G4H * 4);
  float* Bs1f = (float*)alloc((size_t)G4H * 4);
  float* Bs1b = (float*)alloc((size_t)G4H * 4);
  bf16* GXf   = (bf16*)alloc((size_t)Sm * Bm * G4H * 2);     // 67.1 MB
  bf16* GXb   = (bf16*)alloc((size_t)Sm * Bm * G4H * 2);     // 67.1 MB
  bf16* H1    = (bf16*)alloc((size_t)Bm * Sm * D1 * 2);      // 33.6 MB
  bf16* H2    = (bf16*)alloc((size_t)Bm * Sm * D1 * 2);      // 33.6 MB

  // 1) embedding gather + cast
  {
    long long total = (long long)Bm * Sm * E_PAD;
    embed_gather_kernel<<<(unsigned)(total / 256), 256, 0, stream>>>(ids, emb, X0);
  }
  // 2) weight casts + bias sums
  auto cast_w = [&](const float* src, bf16* dst, int K, int Kp) {
    long long total = (long long)G4H * Kp;
    cast_weight_kernel<<<(unsigned)((total + 255) / 256), 256, 0, stream>>>(src, dst, K, Kp, total);
  };
  cast_w(w_ih[0], Wi0f, Em, E_PAD);
  cast_w(w_ih[1], Wi0b, Em, E_PAD);
  cast_w(w_ih[2], Wi1f, D1, D1);
  cast_w(w_ih[3], Wi1b, D1, D1);
  cast_w(w_hh[0], Wh0f, Hm, Hm);
  cast_w(w_hh[1], Wh0b, Hm, Hm);
  cast_w(w_hh[2], Wh1f, Hm, Hm);
  cast_w(w_hh[3], Wh1b, Hm, Hm);
  bias_sum_kernel<<<4, 256, 0, stream>>>(b_ih[0], b_hh[0], Bs0f);
  bias_sum_kernel<<<4, 256, 0, stream>>>(b_ih[1], b_hh[1], Bs0b);
  bias_sum_kernel<<<4, 256, 0, stream>>>(b_ih[2], b_hh[2], Bs1f);
  bias_sum_kernel<<<4, 256, 0, stream>>>(b_ih[3], b_hh[3], Bs1b);

  const dim3 ggrid(G4H / 64, (Bm * Sm) / 128);  // 16 x 256
  const size_t scan_dyn_lds = (size_t)2 * 16 * 64 * 64 * sizeof(bf16);  // 256 KB
  (void)hipFuncSetAttribute((const void*)lstm_scan_kernel,
                            hipFuncAttributeMaxDynamicSharedMemorySize,
                            (int)scan_dyn_lds);

  // 3) layer 0: input projections, then scan
  gemm_gates_kernel<<<ggrid, 256, 0, stream>>>(X0, Wi0f, Bs0f, GXf, E_PAD);
  gemm_gates_kernel<<<ggrid, 256, 0, stream>>>(X0, Wi0b, Bs0b, GXb, E_PAD);
  lstm_scan_kernel<<<2, 512, scan_dyn_lds, stream>>>(GXf, GXb, Wh0f, Wh0b, H1);

  // 4) layer 1: input projections on H1 (rows m = b*S+s), then scan
  gemm_gates_kernel<<<ggrid, 256, 0, stream>>>(H1, Wi1f, Bs1f, GXf, D1);
  gemm_gates_kernel<<<ggrid, 256, 0, stream>>>(H1, Wi1b, Bs1b, GXb, D1);
  lstm_scan_kernel<<<2, 512, scan_dyn_lds, stream>>>(GXf, GXb, Wh1f, Wh1b, H2);

  // 5) head + loss
  zero_loss_kernel<<<1, 64, 0, stream>>>(out);
  head_loss_kernel<<<Bm, 256, 0, stream>>>(H2, w_lin, b_lin, labels, masks, out);
}
